// VAE_50002009260736
// MI455X (gfx1250) — matrix operand order repaired and checked by reading, not compiled, not run on hardware
//
#include <hip/hip_runtime.h>
#include <hip/hip_bf16.h>

// ---------------------------------------------------------------------------
// Masked 2-encoder LSTM (trajectory VAE encoder) on MI455X / gfx1250.
//
// ~245 GFLOP of GEMM vs ~11 MB real HBM traffic -> matrix-pipe bound; all
// GEMMs (gates AND Hidden2Normal) run on v_wmma_f32_16x16x32_bf16.
// Recurrent state lives in LDS per 64-track workgroup (~149 KB of 320 KB/WGP):
//   h  : bf16, row-major, double-buffered (A-operand for both GEMMs)
//   c  : f32, stored in WMMA-accumulator fragment layout (vectorized ds I/O)
// Register blocking: each wave holds 4 M-tiles x 4 gates of f32 accumulators
// (128 VGPRs) so every weight B-fragment is loaded from L2 exactly once per
// wave per step (4x less L2 traffic than naive tiling).
// ---------------------------------------------------------------------------

typedef __attribute__((ext_vector_type(16))) __bf16 v16bf;
typedef __attribute__((ext_vector_type(8)))  float  v8f;

#define EMBD 128
#define HIDD 256
#define GATW 1024          // 4*HID
#define ROWS 64            // tracks per workgroup
#define NSEQ 19            // 8 (enc1) + 11 (enc2) output steps

union FragAB {
  v16bf v;
  uint4 u[2];
};

__device__ __forceinline__ float sigf(float x) { return 1.f / (1.f + __expf(-x)); }

// A operand (16x32 bf16) from row-major LDS buffer, contiguous-K layout:
// lanes 0-15 row M=lane, K = k0+{0..15}; lanes 16-31 row M=lane-16,
// K = k0+{16..31}.  Two 16B ds_load_b128 per lane.
__device__ __forceinline__ v16bf load_a_frag(const __bf16* buf, int kdim,
                                             int mtile, int k0, int lane) {
  int row = mtile * 16 + (lane & 15);
  int klo = k0 + ((lane & 16) ? 16 : 0);
  const __bf16* p = buf + row * kdim + klo;
  FragAB f;
  f.u[0] = *(const uint4*)(p);
  f.u[1] = *(const uint4*)(p + 8);
  return f.v;
}

// B operand (32x16 bf16) from pre-packed global weights: fragment block is
// 32 lanes x 16 contiguous bf16 (32 B per lane = two global_load_b128).
__device__ __forceinline__ v16bf load_b_frag(const __bf16* frag_base, int lane) {
  const uint4* p = (const uint4*)frag_base + 2 * lane;
  FragAB f;
  f.u[0] = p[0];
  f.u[1] = p[1];
  return f.v;
}

// Pack W [G rows x K cols] f32 (rows >= Gvalid zero-filled) into WMMA-B
// fragments of W^T (K x G):
// dst[((kt*G/16 + nt)*32 + lane)*16 + j] =
//     W[nt*16 + (lane&15)][kt*32 + ((lane>>4)&1)*16 + j]
__global__ void pack_weights(const float* __restrict__ W, __bf16* __restrict__ dst,
                             int K, int G, int Gvalid) {
  int idx = blockIdx.x * blockDim.x + threadIdx.x;
  if (idx >= K * G) return;
  int j    = idx & 15;
  int lane = (idx >> 4) & 31;
  int rem  = idx >> 9;
  int ntc  = G >> 4;
  int nt   = rem % ntc;
  int kt   = rem / ntc;
  int n = nt * 16 + (lane & 15);
  int k = kt * 32 + (((lane >> 4) & 1) << 4) + j;
  dst[idx] = (__bf16)((n < Gvalid) ? W[n * K + k] : 0.f);
}

struct Smem {
  float  velx[ROWS], vely[ROWS], o2x[ROWS], o2y[ROWS];
  int    mask[ROWS];
  float  bsum[GATW];                 // bih + bhh, folded once
  __bf16 emb[ROWS * EMBD];           // row-major bf16
  __bf16 h[2][ROWS * HIDD];          // double-buffered recurrent state
  float  c[ROWS * HIDD];             // cell state, ACC-fragment layout:
                                     //   [(mtile*16+nt)*32 + lane][8]
};

__global__ __launch_bounds__(256, 1)
void vae_lstm_kernel(const float* __restrict__ observed,
                     const float* __restrict__ pred_truth,
                     const float* __restrict__ We,  const float* __restrict__ be,
                     const float* __restrict__ bih1, const float* __restrict__ bhh1,
                     const float* __restrict__ bih2, const float* __restrict__ bhh2,
                     const float* __restrict__ bn,
                     const __bf16* __restrict__ wpack,
                     float* __restrict__ out, int N) {
  __shared__ Smem sm;

  const int tid  = threadIdx.x;
  const int lane = tid & 31;
  const int wave = tid >> 5;
  const int enc  = blockIdx.y;           // 0: encoder1 (obs), 1: encoder2 (teacher-forced)
  const int n0   = blockIdx.x * ROWS;

  const int t1 = GATW * EMBD;            // packed Wih element count
  const int t2 = GATW * HIDD;            // packed Whh element count
  const __bf16* wih = wpack + (enc ? (t1 + t2) : 0);
  const __bf16* whh = wpack + (enc ? (2 * t1 + t2) : t1);
  const __bf16* wnp = wpack + 2 * (t1 + t2);   // packed padded Wn^T (8 fragments)
  const float*  bih = enc ? bih2 : bih1;
  const float*  bhh = enc ? bhh2 : bhh1;

  float* out5 = out;                                     // [19, N, 5]
  float* outp = out + (size_t)NSEQ * N * 5;              // [19, N, 2]

  // ---- init: fold biases, zero recurrent state ----
  for (int i = tid; i < GATW; i += 256) sm.bsum[i] = bih[i] + bhh[i];
  for (int i = tid; i < ROWS * HIDD; i += 256) {
    sm.c[i] = 0.f;
    sm.h[0][i] = (__bf16)0.f;
  }
  __syncthreads();

  const int nsteps = enc ? 11 : 8;
  for (int s = 0; s < nsteps; ++s) {
    // frame selection
    const float* f1;
    const float* f2;
    if (enc == 0) {
      f1 = observed + (size_t)s * N * 2;
      f2 = observed + (size_t)(s + 1) * N * 2;
    } else {
      f1 = (s == 0) ? observed + (size_t)8 * N * 2
                    : pred_truth + (size_t)(s - 1) * N * 2;
      f2 = pred_truth + (size_t)s * N * 2;
    }
    const int row19 = enc ? (8 + s) : s;

    // ---- phase A1: velocities + mask ----
    if (tid < ROWS) {
      int ng = n0 + tid;
      float a1x = f1[2 * ng], a1y = f1[2 * ng + 1];
      float a2x = f2[2 * ng], a2y = f2[2 * ng + 1];
      bool m = !((a1x != a1x) || (a2x != a2x));          // NaN mask on x coord
      sm.mask[tid] = m;
      sm.o2x[tid] = a2x;
      sm.o2y[tid] = a2y;
      sm.velx[tid] = m ? (a2x - a1x) : 0.f;
      sm.vely[tid] = m ? (a2y - a1y) : 0.f;
    }
    __syncthreads();

    // ---- phase A2: emb = relu(4*vel @ We^T + be), bf16, 2 elems/thread/iter ----
    for (int i2 = tid; i2 < ROWS * EMBD / 2; i2 += 256) {
      int r = i2 >> 6, e = (i2 & 63) * 2;
      float vx = 4.f * sm.velx[r], vy = 4.f * sm.vely[r];
      float v0 = vx * We[2 * e]     + vy * We[2 * e + 1] + be[e];
      float v1 = vx * We[2 * e + 2] + vy * We[2 * e + 3] + be[e + 1];
      __bf16* d = sm.emb + r * EMBD + e;
      d[0] = (__bf16)fmaxf(v0, 0.f);
      d[1] = (__bf16)fmaxf(v1, 0.f);
    }
    __syncthreads();

    // ---- phase B: WMMA gates + in-register LSTM cell update ----
    const __bf16* hr = sm.h[s & 1];
    __bf16*       hw = sm.h[(s & 1) ^ 1];

    for (int half = 0; half < 2; ++half) {
      const int nt = wave + half * 8;                    // hidden-unit tile 0..15

      v8f acc[4][4];                                     // [mtile][gate]
#pragma unroll
      for (int g = 0; g < 4; ++g) {
        float b = sm.bsum[(g * 16 + nt) * 16 + (lane & 15)];
#pragma unroll
        for (int m = 0; m < 4; ++m)
#pragma unroll
          for (int v = 0; v < 8; ++v) acc[m][g][v] = b;
      }

      // K = 128 from embedding: load each B fragment once, 4 WMMAs against it
#pragma unroll
      for (int kt = 0; kt < 4; ++kt) {
        v16bf a[4];
#pragma unroll
        for (int m = 0; m < 4; ++m) a[m] = load_a_frag(sm.emb, EMBD, m, kt * 32, lane);
#pragma unroll
        for (int g = 0; g < 4; ++g) {
          v16bf b = load_b_frag(wih + ((size_t)(kt * 64 + (g * 16 + nt)) * 512), lane);
#pragma unroll
          for (int m = 0; m < 4; ++m)
            acc[m][g] = __builtin_amdgcn_wmma_f32_16x16x32_bf16(
                false, a[m], false, b, (short)0, acc[m][g], false, false);
        }
      }
      // K = 256 from recurrent h
#pragma unroll
      for (int kt = 0; kt < 8; ++kt) {
        v16bf a[4];
#pragma unroll
        for (int m = 0; m < 4; ++m) a[m] = load_a_frag(hr, HIDD, m, kt * 32, lane);
#pragma unroll
        for (int g = 0; g < 4; ++g) {
          v16bf b = load_b_frag(whh + ((size_t)(kt * 64 + (g * 16 + nt)) * 512), lane);
#pragma unroll
          for (int m = 0; m < 4; ++m)
            acc[m][g] = __builtin_amdgcn_wmma_f32_16x16x32_bf16(
                false, a[m], false, b, (short)0, acc[m][g], false, false);
        }
      }

      // i/f/g/o for hidden unit j sit in the same (lane, vgpr) slot.
      // C layout: element v -> row = mtile*16 + v + 8*(lane>=16), col = lane&15.
#pragma unroll
      for (int m = 0; m < 4; ++m) {
        const int rbase = m * 16 + ((lane & 16) ? 8 : 0);
        const int j     = nt * 16 + (lane & 15);
        float* cp = sm.c + ((m * 16 + nt) * 32 + lane) * 8;   // fragment layout
#pragma unroll
        for (int v = 0; v < 8; ++v) {
          int r = rbase + v;
          float iv = sigf(acc[m][0][v]);
          float fv = sigf(acc[m][1][v]);
          float gv = tanhf(acc[m][2][v]);
          float ov = sigf(acc[m][3][v]);
          float cold = cp[v];
          float cnew = fv * cold + iv * gv;
          float hnew = ov * tanhf(cnew);
          bool  msk  = sm.mask[r] != 0;
          cp[v] = msk ? cnew : cold;
          float hold = (float)hr[r * HIDD + j];
          hw[r * HIDD + j] = (__bf16)(msk ? hnew : hold);
        }
      }
    }
    __syncthreads();

    // ---- phase C: Hidden2Normal via WMMA (h_new @ Wn^T, 5 cols padded to 16) ----
    if (wave < 4) {
      const int m = wave;                 // one M-tile per wave
      const int q = lane & 15;            // output column (0..4 valid)
      v8f acc;
      float binit = (q < 5) ? bn[q] : 0.f;
#pragma unroll
      for (int v = 0; v < 8; ++v) acc[v] = binit;
#pragma unroll
      for (int kt = 0; kt < 8; ++kt) {
        v16bf a = load_a_frag(hw, HIDD, m, kt * 32, lane);
        v16bf b = load_b_frag(wnp + (size_t)kt * 512, lane);
        acc = __builtin_amdgcn_wmma_f32_16x16x32_bf16(
            false, a, false, b, (short)0, acc, false, false);
      }
      if (q < 5) {
        const int rbase = m * 16 + ((lane & 16) ? 8 : 0);
#pragma unroll
        for (int v = 0; v < 8; ++v) {
          int r  = rbase + v;
          int ng = n0 + r;
          float raw = acc[v];
          float val;
          if (q < 2)      val = raw;
          else if (q < 4) val = 0.01f + 0.2f * sigf(raw);
          else            val = 0.7f * sigf(raw);
          bool  msk = sm.mask[r] != 0;
          float o   = msk ? val : __builtin_nanf("");
          out5[((size_t)row19 * N + ng) * 5 + q] = o;
          if (q < 2) {
            float base = (q == 0) ? sm.o2x[r] : sm.o2y[r];
            outp[((size_t)row19 * N + ng) * 2 + q] = base + o;
          }
        }
      }
    }
    __syncthreads();
  }
}

extern "C" void kernel_launch(void* const* d_in, const int* in_sizes, int n_in,
                              void* d_out, int out_size, void* d_ws, size_t ws_size,
                              hipStream_t stream) {
  const float* observed = (const float*)d_in[0];
  // d_in[1] goals, d_in[3] batch_split: unused by the reference computation.
  const float* pred  = (const float*)d_in[2];
  const float* We    = (const float*)d_in[4];
  const float* be    = (const float*)d_in[5];
  const float* Wih1  = (const float*)d_in[6];
  const float* Whh1  = (const float*)d_in[7];
  const float* bih1  = (const float*)d_in[8];
  const float* bhh1  = (const float*)d_in[9];
  const float* Wih2  = (const float*)d_in[10];
  const float* Whh2  = (const float*)d_in[11];
  const float* bih2  = (const float*)d_in[12];
  const float* bhh2  = (const float*)d_in[13];
  const float* Wn    = (const float*)d_in[14];
  const float* bn    = (const float*)d_in[15];

  const int N = in_sizes[1] / 2;          // goals is [N, 2]

  // workspace (~1.6 MB bf16): Wih1 | Whh1 | Wih2 | Whh2 | Wn(padded to 16 rows)
  __bf16* wp = (__bf16*)d_ws;
  const int t1 = GATW * EMBD;             // 131072
  const int t2 = GATW * HIDD;             // 262144
  const int tn = HIDD * 16;               // 4096
  pack_weights<<<(t1 + 255) / 256, 256, 0, stream>>>(Wih1, wp,                 EMBD, GATW, GATW);
  pack_weights<<<(t2 + 255) / 256, 256, 0, stream>>>(Whh1, wp + t1,            HIDD, GATW, GATW);
  pack_weights<<<(t1 + 255) / 256, 256, 0, stream>>>(Wih2, wp + t1 + t2,       EMBD, GATW, GATW);
  pack_weights<<<(t2 + 255) / 256, 256, 0, stream>>>(Whh2, wp + 2 * t1 + t2,   HIDD, GATW, GATW);
  pack_weights<<<(tn + 255) / 256, 256, 0, stream>>>(Wn,   wp + 2 * (t1 + t2), HIDD, 16,   5);

  dim3 grid(N / ROWS, 2);                 // 256 track-tiles x 2 encoders
  vae_lstm_kernel<<<grid, 256, 0, stream>>>(observed, pred, We, be,
                                            bih1, bhh1, bih2, bhh2,
                                            bn, wp, (float*)d_out, N);
}
